// ScaledDotProductAttention_65008624992835
// MI455X (gfx1250) — matrix operand order repaired and checked by reading, not compiled, not run on hardware
//
#include <hip/hip_runtime.h>

// ---------------------------------------------------------------------------
// MI455X (gfx1250, wave32) fused Hawkes-prior attention.
//   Kernel 1: per-(b,i) row: ts*log(softmax_t+eps) + ss*log(softmax_s+eps)
//             into d_ws (B*S*S f32 = 32 MB) -- shared across all 8 heads.
//   Kernel 2: per-(b,h,64-row block): TDM pre-loads the 64x1024 prior panel
//             into LDS (tensor_load_to_lds, TENSORcnt), S = (Q/8)K^T + prior
//             via WMMA f16, row softmax fully in 320KB LDS, write attn_weights,
//             O = P*V via WMMA f16.
// ---------------------------------------------------------------------------

typedef __attribute__((ext_vector_type(16))) _Float16     v16h;
typedef __attribute__((ext_vector_type(8)))  _Float16     v8h;
typedef __attribute__((ext_vector_type(8)))  float        v8f;
typedef __attribute__((ext_vector_type(4)))  unsigned int v4u;
typedef __attribute__((ext_vector_type(8)))  int          v8i;
typedef __attribute__((ext_vector_type(4)))  int          v4i;

#define NB 8
#define NH 8
#define NS 1024
#define ND 64

__device__ __forceinline__ float wave_max32(float v) {
#pragma unroll
  for (int m = 16; m >= 1; m >>= 1) v = fmaxf(v, __shfl_xor(v, m, 32));
  return v;
}
__device__ __forceinline__ float wave_sum32(float v) {
#pragma unroll
  for (int m = 16; m >= 1; m >>= 1) v += __shfl_xor(v, m, 32);
  return v;
}

// TDM: arity differs between ROCm 7.2 (clang-22, 5 args) and therock (clang-23, 6 args)
__device__ __forceinline__ void tdm_load_to_lds(v4u g0, v8i g1) {
  v4i z4 = {0, 0, 0, 0};
#if __clang_major__ >= 23
  v8i z8 = {0, 0, 0, 0, 0, 0, 0, 0};
  __builtin_amdgcn_tensor_load_to_lds(g0, g1, z4, z4, z8, 0);
#else
  __builtin_amdgcn_tensor_load_to_lds(g0, g1, z4, z4, 0);
#endif
}

// ------------------------- Kernel 1: Hawkes priors -------------------------
// grid = B*S/8 blocks of 256 threads; one wave32 per (b,i) row.
__global__ __launch_bounds__(256) void hawkes_prior_kernel(
    const float* __restrict__ et, const float* __restrict__ el,
    const float* __restrict__ td_p, const float* __restrict__ sg_p,
    const float* __restrict__ ts_p, const float* __restrict__ ss_p,
    float* __restrict__ Pz)
{
  const int lane = threadIdx.x & 31;
  const int wave = threadIdx.x >> 5;
  const int gr   = blockIdx.x * 8 + wave;   // 0..B*S-1
  const int b    = gr >> 10;
  const int i    = gr & (NS - 1);

  const float td     = log1pf(expf(td_p[0])) + 1e-4f;     // softplus + 1e-4
  const float sg     = log1pf(expf(sg_p[0])) + 0.1f;      // softplus + 0.1
  const float inv2s2 = 1.0f / (2.0f * sg * sg);
  const float ts     = 1.0f / (1.0f + expf(-ts_p[0]));
  const float ss     = 1.0f / (1.0f + expf(-ss_p[0]));

  const float ti = et[b * NS + i];
  const float lx = el[(b * NS + i) * 2 + 0];
  const float ly = el[(b * NS + i) * 2 + 1];

  float ut[32], us[32];
  float mt = -1e30f, ms = -1e30f;
#pragma unroll
  for (int t = 0; t < 32; ++t) {
    const int j = t * 32 + lane;
    const float dt = fmaxf(et[b * NS + j] - ti, 0.0f);
    const float tb = expf(-td * dt);
    const float te = 0.1f / (1.0f + expf(0.1f * dt)) + 1.0f;   // sigmoid(-0.1dt)*0.1+1
    ut[t] = tb * te * 10.0f;                                   // /PRIOR_TEMP
    mt = fmaxf(mt, ut[t]);

    const float dx = el[(b * NS + j) * 2 + 0] - lx;
    const float dy = el[(b * NS + j) * 2 + 1] - ly;
    const float d2 = dx * dx + dy * dy;
    const float sb = expf(-d2 * inv2s2);
    const float se = 0.1f / (1.0f + expf(0.1f * sqrtf(d2))) + 1.0f;
    us[t] = sb * se * 10.0f;
    ms = fmaxf(ms, us[t]);
  }
  mt = wave_max32(mt);
  ms = wave_max32(ms);
  float zt = 0.0f, zs = 0.0f;
#pragma unroll
  for (int t = 0; t < 32; ++t) { zt += expf(ut[t] - mt); zs += expf(us[t] - ms); }
  zt = wave_sum32(zt);
  zs = wave_sum32(zs);
  const float izt = 1.0f / zt, izs = 1.0f / zs;

  float* out = Pz + (size_t)gr * NS;
#pragma unroll
  for (int t = 0; t < 32; ++t) {
    const float pt = expf(ut[t] - mt) * izt;
    const float ps = expf(us[t] - ms) * izs;
    out[t * 32 + lane] = ts * logf(pt + 1e-8f) + ss * logf(ps + 1e-8f);
  }
}

// --------------------- Kernel 2: fused attention (WMMA) --------------------
// grid = B*H*(S/64) blocks of 256 threads (8 waves).
// Dynamic LDS: 64x1024 f32 logits (256KB, TDM-preloaded with prior)
//            + 64x64 f16 Q (8KB) + 64x64 f16 K/V (8KB).
__global__ __launch_bounds__(256) void fused_attn_kernel(
    const float* __restrict__ q, const float* __restrict__ k,
    const float* __restrict__ v, const float* __restrict__ Pz,
    float* __restrict__ outO, float* __restrict__ outW)
{
  extern __shared__ char smem[];
  float*    Pl  = (float*)smem;                         // [64][1024] prior -> logits -> probs
  _Float16* Qh  = (_Float16*)(smem + 64 * 1024 * 4);    // [64 rows][64 d]
  _Float16* KVh = Qh + 64 * 64;                         // K: [d][j] (transposed); V: [j][d]

  const int tid  = threadIdx.x;
  const int lane = tid & 31;
  const int wave = tid >> 5;
  const int bid  = blockIdx.x;
  const int rb   = bid & 15;
  const int h    = (bid >> 4) & 7;
  const int b    = bid >> 7;
  const int r0   = rb * 64;

  const float* Qg = q + ((size_t)(b * NH + h) * NS) * ND;
  const float* Kg = k + ((size_t)(b * NH + h) * NS) * ND;
  const float* Vg = v + ((size_t)(b * NH + h) * NS) * ND;

  // ---- TDM: async-copy the 64x1024 f32 prior panel into Pl (one op) ------
  if (wave == 0) {
    const unsigned long long ga =
        (unsigned long long)(uintptr_t)(Pz + (size_t)b * NS * NS + (size_t)r0 * NS);
    const unsigned lds_off = (unsigned)(uintptr_t)Pl;   // byte offset in LDS
    v4u g0;
    g0[0] = 1u;                                          // count=1, no gather
    g0[1] = lds_off;                                     // lds_addr
    g0[2] = (unsigned)(ga & 0xFFFFFFFFu);                // global_addr[31:0]
    g0[3] = (unsigned)((ga >> 32) & 0x01FFFFFFu)         // global_addr[56:32]
            | 0x80000000u;                               // type=2 ("image")
    v8i g1;
    g1[0] = 0x00020000;            // wg_mask=0, data_size=2 (4B), no flags
    g1[1] = (int)(1024u << 16);    // tensor_dim0[15:0]=1024 in [31:16]
    g1[2] = (int)(1024u << 16);    // dim0 hi=0; tensor_dim1[15:0]=1024 in [31:16]
    g1[3] = (int)(1024u << 16);    // dim1 hi=0; tile_dim0=1024 in [31:16]
    g1[4] = 64;                    // tile_dim1=64, tile_dim2=0
    g1[5] = 1024;                  // tensor_dim0_stride[31:0]=1024
    g1[6] = 0;                     // stride hi, dim1_stride lo
    g1[7] = 0;
    tdm_load_to_lds(g0, g1);
  }

  // ---- stage Q tile (fold 1/temperature=1/8 into the f16 conversion) ----
  {
    const int row = tid >> 2, d0 = (tid & 3) * 16;
    const float4* src = (const float4*)(Qg + (size_t)(r0 + row) * ND + d0);
#pragma unroll
    for (int p = 0; p < 4; ++p) {
      float4 x = src[p];
      _Float16* dst = Qh + row * ND + d0 + p * 4;
      dst[0] = (_Float16)(x.x * 0.125f);
      dst[1] = (_Float16)(x.y * 0.125f);
      dst[2] = (_Float16)(x.z * 0.125f);
      dst[3] = (_Float16)(x.w * 0.125f);
    }
  }
  if (wave == 0) __builtin_amdgcn_s_wait_tensorcnt(0);   // prior panel in LDS
  __syncthreads();

  const int rt    = wave & 3;          // row tile 0..3 (16 rows each)
  const int cpair = (wave >> 2) * 2;   // this wave's pair of col/d tiles
  const int lrow  = lane & 15;
  const int k0    = (lane < 16) ? 0 : 8;   // A-frag K-offset per ISA layout

  // ================= S = (Q/8) K^T + prior (in-LDS accumulate) =============
  for (int cb = 0; cb < 16; ++cb) {
    const int c0 = cb * 64;
    // load K block transposed into KVh[d][j]  -> contiguous B-fragments
    {
      const int row = tid >> 2, d0 = (tid & 3) * 16;
      const float4* src = (const float4*)(Kg + (size_t)(c0 + row) * ND + d0);
#pragma unroll
      for (int p = 0; p < 4; ++p) {
        float4 x = src[p];
        KVh[(d0 + p * 4 + 0) * 64 + row] = (_Float16)x.x;
        KVh[(d0 + p * 4 + 1) * 64 + row] = (_Float16)x.y;
        KVh[(d0 + p * 4 + 2) * 64 + row] = (_Float16)x.z;
        KVh[(d0 + p * 4 + 3) * 64 + row] = (_Float16)x.w;
      }
      if (cb < 15)   // overlap next K panel fetch with this stage's WMMAs
        __builtin_prefetch(Kg + (size_t)(c0 + 64 + row) * ND + d0, 0, 1);
    }
    __syncthreads();

#pragma unroll
    for (int cc = 0; cc < 2; ++cc) {
      const int ct = cpair + cc;
      v8f acc = {};
#pragma unroll
      for (int kk = 0; kk < 2; ++kk) {   // d = 0..31, 32..63
        // A: lane<16 holds K {0-7,16-23}, lane>=16 holds K {8-15,24-31}
        const _Float16* ap = Qh + (rt * 16 + lrow) * ND + kk * 32 + k0;
        v8h alo = *(const v8h*)ap;
        v8h ahi = *(const v8h*)(ap + 16);
        v16h a;
#pragma unroll
        for (int e = 0; e < 8; ++e) { a[e] = alo[e]; a[8 + e] = ahi[e]; }
        // B: lane%16 = N (j), elements over K (d) in same order as A
        const _Float16* bp = KVh + (kk * 32 + k0) * 64 + ct * 16 + lrow;
        v16h bb;
#pragma unroll
        for (int e = 0; e < 8; ++e) { bb[e] = bp[e * 64]; bb[8 + e] = bp[(16 + e) * 64]; }
        acc = __builtin_amdgcn_wmma_f32_16x16x32_f16(false, a, false, bb,
                                                     (short)0, acc, false, false);
      }
      // D layout: VGPR r -> M = r (lanes<16) / r+8 (lanes>=16), N = lane%16
      // Pl already holds the TDM-loaded prior: accumulate in place.
#pragma unroll
      for (int r = 0; r < 8; ++r) {
        const int row_l = rt * 16 + r + ((lane >> 4) << 3);
        const int idx   = row_l * NS + c0 + ct * 16 + lrow;
        Pl[idx] = acc[r] + Pl[idx];
      }
    }
    __syncthreads();
  }

  // =================== row softmax over LDS, emit weights ==================
  for (int r = wave; r < 64; r += 8) {
    float* row = Pl + r * NS;
    float m = -1e30f;
#pragma unroll
    for (int t = 0; t < 32; ++t) m = fmaxf(m, row[t * 32 + lane]);
    m = wave_max32(m);
    float z = 0.0f;
#pragma unroll
    for (int t = 0; t < 32; ++t) z += expf(row[t * 32 + lane] - m);
    z = wave_sum32(z);
    const float iz = 1.0f / z;
    float* wout = outW + ((size_t)(b * NH + h) * NS + (r0 + r)) * NS;
#pragma unroll
    for (int t = 0; t < 32; ++t) {
      const float p = expf(row[t * 32 + lane] - m) * iz;
      row[t * 32 + lane]  = p;     // keep in LDS for P*V
      wout[t * 32 + lane] = p;     // coalesced 128B per step
    }
  }
  __syncthreads();

  // ============================ O = P * V ==================================
  v8f o0 = {}, o1 = {};
  for (int jb = 0; jb < 16; ++jb) {
    __syncthreads();               // all waves done reading previous KVh block
    {
      const int row = tid >> 2, d0 = (tid & 3) * 16;
      const float4* src = (const float4*)(Vg + (size_t)(jb * 64 + row) * ND + d0);
#pragma unroll
      for (int p = 0; p < 4; ++p) {
        float4 x = src[p];
        _Float16* dst = KVh + row * ND + d0 + p * 4;   // V kept [j][d]
        dst[0] = (_Float16)x.x; dst[1] = (_Float16)x.y;
        dst[2] = (_Float16)x.z; dst[3] = (_Float16)x.w;
      }
      if (jb < 15)
        __builtin_prefetch(Vg + (size_t)(jb * 64 + 64 + row) * ND + d0, 0, 1);
    }
    __syncthreads();

#pragma unroll
    for (int kk = 0; kk < 2; ++kk) {  // j = 0..31, 32..63 within block
      const float* prow = Pl + (rt * 16 + lrow) * NS + jb * 64 + kk * 32 + k0;
      v16h a;
#pragma unroll
      for (int e = 0; e < 8; ++e) {
        a[e]     = (_Float16)prow[e];
        a[8 + e] = (_Float16)prow[16 + e];
      }
      const _Float16* bp = KVh + (kk * 32 + k0) * 64 + lrow;   // lane%16 = N (d), elems over K (j)
      v16h b0, b1;
#pragma unroll
      for (int e = 0; e < 8; ++e) {
        b0[e] = bp[e * 64 + cpair * 16]; b0[8 + e] = bp[(16 + e) * 64 + cpair * 16];
        b1[e] = bp[e * 64 + (cpair + 1) * 16]; b1[8 + e] = bp[(16 + e) * 64 + (cpair + 1) * 16];
      }
      o0 = __builtin_amdgcn_wmma_f32_16x16x32_f16(false, a, false, b0,
                                                  (short)0, o0, false, false);
      o1 = __builtin_amdgcn_wmma_f32_16x16x32_f16(false, a, false, b1,
                                                  (short)0, o1, false, false);
    }
  }

  float* Og = outO + ((size_t)(b * NH + h) * NS + r0) * ND;
#pragma unroll
  for (int r = 0; r < 8; ++r) {
    const int row_l = rt * 16 + r + ((lane >> 4) << 3);
    Og[row_l * ND + cpair * 16 + lrow]       = o0[r];
    Og[row_l * ND + (cpair + 1) * 16 + lrow] = o1[r];
  }
}

// ---------------------------------------------------------------------------
extern "C" void kernel_launch(void* const* d_in, const int* in_sizes, int n_in,
                              void* d_out, int out_size, void* d_ws, size_t ws_size,
                              hipStream_t stream) {
  const float* q   = (const float*)d_in[0];
  const float* k   = (const float*)d_in[1];
  const float* v   = (const float*)d_in[2];
  const float* et  = (const float*)d_in[3];
  const float* el  = (const float*)d_in[4];
  const float* td  = (const float*)d_in[5];
  const float* sg  = (const float*)d_in[6];
  const float* tsp = (const float*)d_in[7];
  const float* ssp = (const float*)d_in[8];

  float* Pz   = (float*)d_ws;                          // needs B*S*S*4 = 32 MB
  float* outO = (float*)d_out;                         // (B,H,S,D)
  float* outW = outO + (size_t)NB * NH * NS * ND;      // (B,H,S,S)

  // Kernel 1: one wave per (b,i) row -> B*S/8 = 1024 blocks
  hipLaunchKernelGGL(hawkes_prior_kernel, dim3(NB * NS / 8), dim3(256), 0, stream,
                     et, el, td, sg, tsp, ssp, Pz);

  // Kernel 2: B*H*(S/64) = 1024 blocks, 272 KB dynamic LDS (fits 320KB/WGP)
  const size_t smem = 64 * 1024 * sizeof(float) + 2 * 64 * 64 * sizeof(short);
  hipLaunchKernelGGL(fused_attn_kernel, dim3(NB * NH * (NS / 64)), dim3(256),
                     smem, stream, q, k, v, Pz, outO, outW);
}
